// NGNNConv_23184233463958
// MI455X (gfx1250) — matrix operand order repaired and checked by reading, not compiled, not run on hardware
//
#include <hip/hip_runtime.h>

#define NN 128
#define DIN 64
#define DOUT 64

typedef __attribute__((ext_vector_type(8)))  float   v8f;
typedef __attribute__((ext_vector_type(4)))  __bf16  bf16x4;
typedef __attribute__((ext_vector_type(8)))  __bf16  bf16x8;
typedef __attribute__((ext_vector_type(16))) __bf16  bf16x16;

__device__ __forceinline__ bf16x8 ldv8(const unsigned short* p) {
    return *(const bf16x8*)p;                 // 16B-aligned by construction
}
__device__ __forceinline__ bf16x16 cat(bf16x8 a, bf16x8 b) {
    return __builtin_shufflevector(a, b, 0,1,2,3,4,5,6,7,8,9,10,11,12,13,14,15);
}
__device__ __forceinline__ unsigned short bfbits(float f) {
    return __builtin_bit_cast(unsigned short, (__bf16)f);   // native v_cvt, RNE
}

// Fused: out[b,i,j,e] = sum_k A[b,k,j] * ( sum_d X[b,i,k,d]*W[d,e] + bias[e] )
// One workgroup (8 wave32) per (b,i). Split-bf16 WMMA with f32 accumulate.
// Waves tile the 128x64 output as a 4x2 wave grid, each wave owning a 2x2
// block of 16x16 tiles for balanced A/B LDS operand reuse.
__global__ __launch_bounds__(256) void ngnn_fused(
    const float* __restrict__ Xg,   // [B,N,N,DIN]
    const float* __restrict__ Ag,   // [B,N,N]
    const float* __restrict__ Wg,   // [DIN,DOUT]
    const float* __restrict__ bg,   // [DOUT]
    float* __restrict__ outg)       // [B,N,N,DOUT]
{
    extern __shared__ unsigned short smem[];
    unsigned short* sXh = smem;              // 128 x 72  (X hi, row-major [k][d])
    unsigned short* sXl = smem + 9216;       // 128 x 72  (X lo)
    unsigned short* sWh = smem + 18432;      //  64 x 72  (W^T hi, [e][d])
    unsigned short* sWl = smem + 23040;      //  64 x 72  (W^T lo)
    unsigned short* sAt = smem + 27648;      // 128 x 136 (A^T, [j][k], exact 0/1)
    unsigned short* sTh = smem + 45056;      //  64 x 136 (tX^T hi, [e][k])
    unsigned short* sTl = smem + 53760;      //  64 x 136 (tX^T lo)

    const int i    = blockIdx.x;
    const int b    = blockIdx.y;
    const int t    = threadIdx.x;
    const int lane = t & 31;
    const int w    = t >> 5;       // wave id 0..7
    const int mw   = w >> 1;       // wave M position 0..3 (2 tiles each)
    const int nw   = w & 1;        // wave N position 0..1 (2 tiles each)
    const int lr   = lane & 15;    // row/col within 16x16 tile
    const int h    = lane >> 4;    // lane half

    const float* Xs = Xg + (size_t)(b * NN + i) * (NN * DIN);
    const float* As = Ag + (size_t)b * (NN * NN);

    // ---- Stage X slice (128x64 f32) -> split bf16, row-major, stride 72 ----
    const float4* X4 = (const float4*)Xs;
#pragma unroll
    for (int it = 0; it < 8; ++it) {
        int v = t + it * 256;                 // 0..2047
        float4 x = X4[v];
        int k = v >> 4;
        int d = (v & 15) << 2;
        float xs[4] = {x.x, x.y, x.z, x.w};
        bf16x4 hv, lv;
#pragma unroll
        for (int q = 0; q < 4; ++q) {
            __bf16 hi = (__bf16)xs[q];
            hv[q] = hi;
            lv[q] = (__bf16)(xs[q] - (float)hi);
        }
        *(bf16x4*)&sXh[k * 72 + d] = hv;
        *(bf16x4*)&sXl[k * 72 + d] = lv;
    }
    // ---- Stage W (64x64) transposed + split, stride 72 ----
    const float4* W4 = (const float4*)Wg;
#pragma unroll
    for (int it = 0; it < 4; ++it) {
        int v = t + it * 256;                 // 0..1023
        float4 x = W4[v];
        int d = v >> 4;
        int e = (v & 15) << 2;
        float xs[4] = {x.x, x.y, x.z, x.w};
#pragma unroll
        for (int q = 0; q < 4; ++q) {
            __bf16 hi = (__bf16)xs[q];
            sWh[(e + q) * 72 + d] = __builtin_bit_cast(unsigned short, hi);
            sWl[(e + q) * 72 + d] = bfbits(xs[q] - (float)hi);
        }
    }
    // ---- Stage A[b] (128x128) transposed -> bf16 (exact), stride 136 ----
    const float4* A4 = (const float4*)As;
#pragma unroll
    for (int it = 0; it < 16; ++it) {
        int v = t + it * 256;                 // 0..4095
        float4 x = A4[v];
        int k = v >> 5;
        int j = (v & 31) << 2;
        float xs[4] = {x.x, x.y, x.z, x.w};
#pragma unroll
        for (int q = 0; q < 4; ++q)
            sAt[(j + q) * 136 + k] = bfbits(xs[q]);
    }
    __syncthreads();

    // ---- GEMM1: tX(128x64) = X @ W + bias ; store split-bf16 transposed ----
    {
        v8f acc[2][2] = {};
#pragma unroll
        for (int ks = 0; ks < 2; ++ks) {
            bf16x16 ah[2], al[2];
#pragma unroll
            for (int mi = 0; mi < 2; ++mi) {
                int offa = (mw * 32 + mi * 16 + lr) * 72 + ks * 32 + h * 8;
                ah[mi] = cat(ldv8(&sXh[offa]), ldv8(&sXh[offa + 16]));
                al[mi] = cat(ldv8(&sXl[offa]), ldv8(&sXl[offa + 16]));
            }
            bf16x16 bh[2], bl[2];
#pragma unroll
            for (int ni = 0; ni < 2; ++ni) {
                int offb = (nw * 32 + ni * 16 + lr) * 72 + ks * 32 + h * 8;
                bh[ni] = cat(ldv8(&sWh[offb]), ldv8(&sWh[offb + 16]));
                bl[ni] = cat(ldv8(&sWl[offb]), ldv8(&sWl[offb + 16]));
            }
#pragma unroll
            for (int mi = 0; mi < 2; ++mi)
#pragma unroll
                for (int ni = 0; ni < 2; ++ni) {
                    acc[mi][ni] = __builtin_amdgcn_wmma_f32_16x16x32_bf16(false, ah[mi], false, bh[ni], (short)0, acc[mi][ni], false, false);
                    acc[mi][ni] = __builtin_amdgcn_wmma_f32_16x16x32_bf16(false, ah[mi], false, bl[ni], (short)0, acc[mi][ni], false, false);
                    acc[mi][ni] = __builtin_amdgcn_wmma_f32_16x16x32_bf16(false, al[mi], false, bh[ni], (short)0, acc[mi][ni], false, false);
                }
        }
        // epilogue: add bias, split fp32 -> (hi,lo) bf16, store transposed
#pragma unroll
        for (int mi = 0; mi < 2; ++mi)
#pragma unroll
            for (int ni = 0; ni < 2; ++ni) {
                int e = nw * 32 + ni * 16 + lr;
                float bias = bg[e];
                bf16x4 h0, h1, l0, l1;
#pragma unroll
                for (int r = 0; r < 8; ++r) {
                    float f = acc[mi][ni][r] + bias;
                    __bf16 hi = (__bf16)f;
                    __bf16 lo = (__bf16)(f - (float)hi);
                    if (r < 4) { h0[r] = hi; l0[r] = lo; }
                    else       { h1[r - 4] = hi; l1[r - 4] = lo; }
                }
                int off = e * 136 + (mw * 32 + mi * 16) + h * 8;
                *(bf16x4*)&sTh[off]     = h0;
                *(bf16x4*)&sTh[off + 4] = h1;
                *(bf16x4*)&sTl[off]     = l0;
                *(bf16x4*)&sTl[off + 4] = l1;
            }
    }
    __syncthreads();

    // ---- GEMM2: out(128x64) = A^T @ tX  (A exact in bf16; tX split hi+lo) ----
    {
        float* outs = outg + (size_t)(b * NN + i) * (NN * DOUT);
        v8f acc[2][2] = {};
#pragma unroll
        for (int ks = 0; ks < 4; ++ks) {
            bf16x16 am[2];
#pragma unroll
            for (int mi = 0; mi < 2; ++mi) {
                int offA = (mw * 32 + mi * 16 + lr) * 136 + ks * 32 + h * 8;
                am[mi] = cat(ldv8(&sAt[offA]), ldv8(&sAt[offA + 16]));
            }
            bf16x16 th[2], tl[2];
#pragma unroll
            for (int ni = 0; ni < 2; ++ni) {
                int offB = (nw * 32 + ni * 16 + lr) * 136 + ks * 32 + h * 8;
                th[ni] = cat(ldv8(&sTh[offB]), ldv8(&sTh[offB + 16]));
                tl[ni] = cat(ldv8(&sTl[offB]), ldv8(&sTl[offB + 16]));
            }
#pragma unroll
            for (int mi = 0; mi < 2; ++mi)
#pragma unroll
                for (int ni = 0; ni < 2; ++ni) {
                    acc[mi][ni] = __builtin_amdgcn_wmma_f32_16x16x32_bf16(false, am[mi], false, th[ni], (short)0, acc[mi][ni], false, false);
                    acc[mi][ni] = __builtin_amdgcn_wmma_f32_16x16x32_bf16(false, am[mi], false, tl[ni], (short)0, acc[mi][ni], false, false);
                }
        }
#pragma unroll
        for (int mi = 0; mi < 2; ++mi)
#pragma unroll
            for (int ni = 0; ni < 2; ++ni) {
                int e = nw * 32 + ni * 16 + lr;
#pragma unroll
                for (int r = 0; r < 8; ++r) {
                    int j = mw * 32 + mi * 16 + h * 8 + r;
                    outs[(size_t)j * DOUT + e] = acc[mi][ni][r];
                }
            }
    }
}

extern "C" void kernel_launch(void* const* d_in, const int* in_sizes, int n_in,
                              void* d_out, int out_size, void* d_ws, size_t ws_size,
                              hipStream_t stream) {
    (void)in_sizes; (void)n_in; (void)out_size; (void)d_ws; (void)ws_size;
    const float* X    = (const float*)d_in[0];
    const float* A    = (const float*)d_in[1];
    const float* W    = (const float*)d_in[2];
    const float* bias = (const float*)d_in[3];
    float* out        = (float*)d_out;

    dim3 grid(NN, 64);   // (i, b) -> 8192 workgroups
    dim3 block(256);     // 8 wave32
    size_t shmem = 124928; // bytes of LDS partitioned above
    hipLaunchKernelGGL(ngnn_fused, grid, block, shmem, stream, X, A, W, bias, out);
}
